// MHSAttention_29308856828428
// MI455X (gfx1250) — matrix operand order repaired and checked by reading, not compiled, not run on hardware
//
#include <hip/hip_runtime.h>

typedef __attribute__((ext_vector_type(8)))  float  v8f;
typedef __attribute__((ext_vector_type(16))) __bf16 v16bf;

#define LSEQ 2048
#define EMB  768
#define NHEAD 12
#define HD   64

static __device__ __forceinline__ v8f wmma_bf16(v16bf a, v16bf b, v8f c) {
  // D = A(16x32 bf16) * B(32x16 bf16) + C(16x16 f32)
  return __builtin_amdgcn_wmma_f32_16x16x32_bf16(false, a, false, b, (short)0, c,
                                                 false, false);
}

// Async DMA: global -> LDS, 16 bytes per lane, tracked by ASYNCcnt.
// LDS offset = low 32 bits of the generic pointer (AS3->flat puts the LDS
// byte offset in the low word).
#define LDS_OFF(p) ((unsigned)(unsigned long long)(p))

static __device__ __forceinline__ void async_ld16(unsigned lds_off, const void* g) {
  asm volatile("global_load_async_to_lds_b128 %0, %1, off"
               :: "v"(lds_off), "v"(g) : "memory");
}

static __device__ __forceinline__ void wait_async0() {
#if __has_builtin(__builtin_amdgcn_s_wait_asynccnt)
  __builtin_amdgcn_s_wait_asynccnt(0);
#else
  asm volatile("s_wait_asynccnt 0x0" ::: "memory");
#endif
}

// ---------------------------------------------------------------------------
// GEMM: out[m,n] = sum_k X[m,k] * W[n,k] + bias[n]    (x @ W^T + b)
// M = 8192, N = K = 768. Block = 256 thr (8 waves), tile 128x128, k-step 32.
// Wave (waveM 0..3, waveN 0..1) owns a 32x64 sub-tile -> 2x4 WMMA frags.
// fp32 inputs convert to bf16 while staging; bf16 inputs stage via async DMA.
// ---------------------------------------------------------------------------
template <typename TIN, typename TOUT>
__global__ __launch_bounds__(256)
void gemm_bias_768(const TIN* __restrict__ X, const float* __restrict__ W,
                   const float* __restrict__ bias, TOUT* __restrict__ out) {
  __shared__ __bf16 As[128 * 32];   // 8 KB
  __shared__ __bf16 Bs[128 * 32];   // 8 KB
  const int tid  = threadIdx.x;
  const int lane = tid & 31, wid = tid >> 5;
  const int waveM = wid >> 1, waveN = wid & 1;
  const int mblk = blockIdx.x * 128, nblk = blockIdx.y * 128;
  const int srow = tid >> 1, sc0 = (tid & 1) * 16;   // staging: 2 thr/row, 16 elems each

  v8f acc[2][4] = {};

  const TIN*   xrow = X + (size_t)(mblk + srow) * EMB + sc0;
  const float* wrow = W + (size_t)(nblk + srow) * EMB + sc0;

  for (int ks = 0; ks < EMB; ks += 32) {
    // ---- stage A tile ----
    if constexpr (sizeof(TIN) == 4) {
      const float4* p = reinterpret_cast<const float4*>(xrow + ks);
#pragma unroll
      for (int q = 0; q < 4; ++q) {
        float4 f = p[q];
        __bf16* d = &As[srow * 32 + sc0 + q * 4];
        d[0] = (__bf16)f.x; d[1] = (__bf16)f.y; d[2] = (__bf16)f.z; d[3] = (__bf16)f.w;
      }
    } else {
      // bf16 input: straight copy -> async DMA to LDS (ASYNCcnt path)
      async_ld16(LDS_OFF(&As[srow * 32 + sc0]),     xrow + ks);
      async_ld16(LDS_OFF(&As[srow * 32 + sc0 + 8]), xrow + ks + 8);
    }
    // ---- stage B tile (weights, fp32 in -> convert) ----
    {
      const float4* p = reinterpret_cast<const float4*>(wrow + ks);
#pragma unroll
      for (int q = 0; q < 4; ++q) {
        float4 f = p[q];
        __bf16* d = &Bs[srow * 32 + sc0 + q * 4];
        d[0] = (__bf16)f.x; d[1] = (__bf16)f.y; d[2] = (__bf16)f.z; d[3] = (__bf16)f.w;
      }
    }
    if constexpr (sizeof(TIN) == 2) wait_async0();
    __syncthreads();

    // ---- fragments ----
    v16bf a[2], b[4];
    const int ko = (lane >> 4) * 8;   // A frag: half-wave K split {0..7,16..23}/{8..15,24..31}
#pragma unroll
    for (int mi = 0; mi < 2; ++mi) {
      const __bf16* s = &As[(waveM * 32 + mi * 16 + (lane & 15)) * 32];
#pragma unroll
      for (int j = 0; j < 8; ++j) { a[mi][j] = s[ko + j]; a[mi][8 + j] = s[ko + 16 + j]; }
    }
    const int kb = (lane >> 4) * 8;   // B frag: lane = column, half-wave K split {0..7,16..23}/{8..15,24..31}
#pragma unroll
    for (int ni = 0; ni < 4; ++ni) {
      const __bf16* s = &Bs[(waveN * 64 + ni * 16 + (lane & 15)) * 32 + kb];
#pragma unroll
      for (int j = 0; j < 8; ++j) { b[ni][j] = s[j]; b[ni][8 + j] = s[16 + j]; }
    }
#pragma unroll
    for (int mi = 0; mi < 2; ++mi) {
#pragma unroll
      for (int ni = 0; ni < 4; ++ni)
        acc[mi][ni] = wmma_bf16(a[mi], b[ni], acc[mi][ni]);
    }
    __syncthreads();
  }

  // ---- epilogue: C layout = (VGPR r -> row half*8+r, lane&15 -> col) ----
  const int col = lane & 15;
  const int rhalf = (lane >> 4) * 8;
#pragma unroll
  for (int ni = 0; ni < 4; ++ni) {
    const int n = nblk + waveN * 64 + ni * 16 + col;
    const float bv = bias[n];
#pragma unroll
    for (int mi = 0; mi < 2; ++mi) {
      const int mrow = mblk + waveM * 32 + mi * 16 + rhalf;
#pragma unroll
      for (int r = 0; r < 8; ++r)
        out[(size_t)(mrow + r) * EMB + n] = (TOUT)(acc[mi][ni][r] + bv);
    }
  }
}

// ---------------------------------------------------------------------------
// Flash attention per (b, h, q-tile of 128 rows). 8 waves x 16 q-rows each.
// K tile staged via async DMA (global -> LDS, ASYNCcnt); V^T staged with a
// transpose through VGPRs; online softmax in fp32.
// ---------------------------------------------------------------------------
__global__ __launch_bounds__(256)
void attn_flash(const __bf16* __restrict__ Q, const __bf16* __restrict__ K,
                const __bf16* __restrict__ V, __bf16* __restrict__ O) {
  __shared__ __bf16 Ks[128 * 64];        // K tile  [key][d]      16 KB
  __shared__ __bf16 Vs[64 * 136];        // V^T tile [d][key]+pad ~17 KB
  __shared__ __bf16 Ps[8 * 16 * 136];    // per-wave P buffer     ~35 KB
  const int tid = threadIdx.x, lane = tid & 31, wid = tid >> 5;
  const int h = blockIdx.y, b = blockIdx.z;
  const int q0 = blockIdx.x * 128 + wid * 16;
  const size_t base = ((size_t)b * LSEQ) * EMB + h * HD;

  // Q fragments for this wave's 16 rows (d split 0..31 / 32..63)
  v16bf aq[2];
  {
    const __bf16* qp = Q + base + (size_t)(q0 + (lane & 15)) * EMB;
    const int ko = (lane >> 4) * 8;
#pragma unroll
    for (int f = 0; f < 2; ++f) {
#pragma unroll
      for (int j = 0; j < 8; ++j) {
        aq[f][j]     = qp[f * 32 + ko + j];
        aq[f][8 + j] = qp[f * 32 + ko + 16 + j];
      }
    }
  }

  v8f o[4] = {};
  float m_i[8], l_i[8];
#pragma unroll
  for (int r = 0; r < 8; ++r) { m_i[r] = -1e30f; l_i[r] = 0.f; }
  const float scale = 0.03608439182435161f;   // 1/sqrt(768), per reference
  __bf16* pw = &Ps[wid * 16 * 136];
  const int skey = tid >> 1, sc0 = (tid & 1) * 32;

  for (int kt = 0; kt < LSEQ; kt += 128) {
    // ---- stage K [128][64] via async DMA; V^T [64][128] via VGPR transpose ----
    {
      const __bf16* kp = K + base + (size_t)(kt + skey) * EMB + sc0;
#pragma unroll
      for (int q4 = 0; q4 < 4; ++q4)
        async_ld16(LDS_OFF(&Ks[skey * 64 + sc0 + q4 * 8]), kp + q4 * 8);
      const __bf16* vp = V + base + (size_t)(kt + skey) * EMB + sc0;
#pragma unroll
      for (int j = 0; j < 32; ++j)
        Vs[(sc0 + j) * 136 + skey] = vp[j];
      if (kt + 128 < LSEQ) {                       // pull next tile toward L2
        __builtin_prefetch(kp + (size_t)128 * EMB, 0, 1);
        __builtin_prefetch(vp + (size_t)128 * EMB, 0, 1);
      }
    }
    wait_async0();
    __syncthreads();

    // ---- S = Q K^T : 8 tiles of 16x16, contraction over d=64 (2 WMMA) ----
    v8f s[8];
    {
      const int kb = (lane >> 4) * 8;
#pragma unroll
      for (int t = 0; t < 8; ++t) {
        v8f accS = {};
#pragma unroll
        for (int f = 0; f < 2; ++f) {
          v16bf bk;
          const __bf16* src = &Ks[(t * 16 + (lane & 15)) * 64 + f * 32 + kb];
#pragma unroll
          for (int j = 0; j < 8; ++j) { bk[j] = src[j]; bk[8 + j] = src[16 + j]; }
          accS = wmma_bf16(aq[f], bk, accS);
        }
        s[t] = accS;
      }
    }

    // ---- online softmax (rows live in VGPR index + lane-half; cols in lanes) ----
    float alpha[8];
#pragma unroll
    for (int r = 0; r < 8; ++r) {
      float mx = -1e30f;
#pragma unroll
      for (int t = 0; t < 8; ++t) { s[t][r] *= scale; mx = fmaxf(mx, s[t][r]); }
#pragma unroll
      for (int off = 1; off < 16; off <<= 1) mx = fmaxf(mx, __shfl_xor(mx, off, 32));
      const float mn = fmaxf(m_i[r], mx);
      alpha[r] = __expf(m_i[r] - mn);
      m_i[r] = mn;
      float rs = 0.f;
#pragma unroll
      for (int t = 0; t < 8; ++t) { float p = __expf(s[t][r] - mn); s[t][r] = p; rs += p; }
#pragma unroll
      for (int off = 1; off < 16; off <<= 1) rs += __shfl_xor(rs, off, 32);
      l_i[r] = l_i[r] * alpha[r] + rs;
    }

    // ---- P (bf16) -> per-wave LDS buffer, then reread in A-frag layout ----
    {
      const int prow = (lane >> 4) * 8;
      const int pcol = lane & 15;
#pragma unroll
      for (int t = 0; t < 8; ++t) {
#pragma unroll
        for (int r = 0; r < 8; ++r)
          pw[(prow + r) * 136 + t * 16 + pcol] = (__bf16)s[t][r];
      }
    }
    asm volatile("s_wait_dscnt 0" ::: "memory");   // wave-local LDS RAW

    // ---- O *= alpha ; O += P * V ----
#pragma unroll
    for (int n = 0; n < 4; ++n) {
#pragma unroll
      for (int r = 0; r < 8; ++r) o[n][r] *= alpha[r];
    }
    {
      const int prow = lane & 15;
      const int ko = (lane >> 4) * 8;
      const int kb = (lane >> 4) * 8;
#pragma unroll
      for (int g = 0; g < 4; ++g) {                // contraction over 128 keys
        v16bf ap;
        const __bf16* src = &pw[prow * 136 + g * 32 + ko];
#pragma unroll
        for (int j = 0; j < 8; ++j) { ap[j] = src[j]; ap[8 + j] = src[16 + j]; }
#pragma unroll
        for (int n = 0; n < 4; ++n) {              // 4 output d-tiles
          v16bf bv;
          const __bf16* vsrc = &Vs[(n * 16 + (lane & 15)) * 136 + g * 32 + kb];
#pragma unroll
          for (int j = 0; j < 8; ++j) { bv[j] = vsrc[j]; bv[8 + j] = vsrc[16 + j]; }
          o[n] = wmma_bf16(ap, bv, o[n]);
        }
      }
    }
    __syncthreads();
  }

  // ---- normalize + write bf16 ----
  {
    const int prow = (lane >> 4) * 8;
    const int pcol = lane & 15;
#pragma unroll
    for (int r = 0; r < 8; ++r) {
      const float inv = 1.0f / l_i[r];
      __bf16* op = O + base + (size_t)(q0 + prow + r) * EMB;
#pragma unroll
      for (int n = 0; n < 4; ++n)
        op[n * 16 + pcol] = (__bf16)(o[n][r] * inv);
    }
  }
}

// ---------------------------------------------------------------------------
extern "C" void kernel_launch(void* const* d_in, const int* in_sizes, int n_in,
                              void* d_out, int out_size, void* d_ws, size_t ws_size,
                              hipStream_t stream) {
  const float* key_in = (const float*)d_in[0];
  const float* query  = (const float*)d_in[1];
  const float* value  = (const float*)d_in[2];
  const float* Wk = (const float*)d_in[3];
  const float* bk = (const float*)d_in[4];
  const float* Wq = (const float*)d_in[5];
  const float* bq = (const float*)d_in[6];
  const float* Wv = (const float*)d_in[7];
  const float* bv = (const float*)d_in[8];
  const float* Wfc = (const float*)d_in[9];
  const float* bfc = (const float*)d_in[10];
  float* out = (float*)d_out;

  const size_t SZ = (size_t)4 * LSEQ * EMB;   // 6,291,456 elems; 4*SZ*2B ~= 48 MB ws
  __bf16* Qb = (__bf16*)d_ws;
  __bf16* Kb = Qb + SZ;
  __bf16* Vb = Kb + SZ;
  __bf16* AO = Vb + SZ;

  dim3 gg(64, 6, 1), bb(256, 1, 1);
  gemm_bias_768<float, __bf16><<<gg, bb, 0, stream>>>(query,  Wq, bq, Qb);
  gemm_bias_768<float, __bf16><<<gg, bb, 0, stream>>>(key_in, Wk, bk, Kb);
  gemm_bias_768<float, __bf16><<<gg, bb, 0, stream>>>(value,  Wv, bv, Vb);
  attn_flash<<<dim3(16, NHEAD, 4), bb, 0, stream>>>(Qb, Kb, Vb, AO);
  gemm_bias_768<__bf16, float><<<gg, bb, 0, stream>>>(AO, Wfc, bfc, out);
}
